// BNNCONV3D_5179730559013
// MI455X (gfx1250) — matrix operand order repaired and checked by reading, not compiled, not run on hardware
//
#include <hip/hip_runtime.h>

// ---- CDNA5 WMMA vector types ----
typedef __attribute__((ext_vector_type(16))) __bf16 v16bf;
typedef __attribute__((ext_vector_type(8)))  __bf16 v8bf;
typedef __attribute__((ext_vector_type(8)))  float  v8f;

// Problem constants: x (8,32,16,160,160) f32, w (32,32,1,3,3) f32, pad (0,(1,1),(1,1))
#define N_   8
#define C_   32
#define D_   16
#define H_   160
#define W_   160
#define HT   4              // output rows per block
#define LDSROWS (HT + 2)    // input rows staged (halo)
#define LDSW    (W_ + 2)    // halo columns
static_assert(LDSROWS * LDSW * C_ * 2 <= 64 * 1024, "LDS budget");

// ---------------------------------------------------------------------------
// Kernel 1: binarize weights.  scale = mean|w| over 3x3, wb[tap][o][c] = bf16(scale*sign(w))
// ---------------------------------------------------------------------------
__global__ __launch_bounds__(256) void bnn_prep_weights(
    const float* __restrict__ w, unsigned short* __restrict__ wb_u) {
  __bf16* wb = (__bf16*)wb_u;
  int oc = blockIdx.x * blockDim.x + threadIdx.x;
  if (oc >= 32 * 32) return;
  int o = oc >> 5, c = oc & 31;
  const float* wp = w + (size_t)(o * 32 + c) * 9;   // w[o][c][0][kh][kw]
  float v[9], s = 0.f;
#pragma unroll
  for (int k = 0; k < 9; ++k) { v[k] = wp[k]; s += fabsf(v[k]); }
  s *= (1.0f / 9.0f);
#pragma unroll
  for (int k = 0; k < 9; ++k) {
    float b = (v[k] > 0.f) ? s : ((v[k] < 0.f) ? -s : 0.f);  // sign(0)=0
    wb[(k * 32 + o) * 32 + c] = (__bf16)b;
  }
}

// ---------------------------------------------------------------------------
// Kernel 2: implicit-GEMM 3x3 conv via v_wmma_f32_16x16x32_bf16.
// Block = 320 threads (10 waves). Block covers one (n,d, 4-row) slab of full W.
// Wave w owns column tile [16w, 16w+16).  9 taps x 2 M-tiles = 18 WMMAs / 16px.
// ---------------------------------------------------------------------------
__global__ __launch_bounds__(320) void bnn_conv3d_wmma(
    const float* __restrict__ x, const unsigned short* __restrict__ wb_u,
    float* __restrict__ out) {
  const __bf16* wb = (const __bf16*)wb_u;
  __shared__ alignas(32) __bf16 xs[LDSROWS * LDSW * C_];  // [row][w+1][c], 62.2 KB

  const int bid = blockIdx.x;
  const int hb  = bid % (H_ / HT);
  const int d   = (bid / (H_ / HT)) % D_;
  const int n   =  bid / ((H_ / HT) * D_);
  const int h0  = hb * HT;
  const int tid = threadIdx.x;

  const size_t cstride = (size_t)D_ * H_ * W_;           // 409600: channel stride
  const float* xbase = x + (size_t)n * C_ * cstride + (size_t)d * (H_ * W_);

  // ---- stage x slab into LDS as bf16, channel-contiguous; w fastest => coalesced ----
  for (int idx = tid; idx < LDSROWS * LDSW * C_; idx += 320) {
    int wl = idx % LDSW;            // 0..161 -> input col wl-1
    int t  = idx / LDSW;
    int c  = t & 31;
    int r  = t >> 5;                // 0..5  -> input row h0-1+r
    int gh = h0 - 1 + r;
    int gw = wl - 1;
    float v = 0.f;
    if ((unsigned)gh < (unsigned)H_ && (unsigned)gw < (unsigned)W_)
      v = xbase[(size_t)c * cstride + gh * W_ + gw];
    xs[(r * LDSW + wl) * C_ + c] = (__bf16)v;
  }
  __syncthreads();

  const int lane = tid & 31;
  const int wave = tid >> 5;            // 0..9 -> column tile
  const int row  = lane & 15;           // A-matrix row / D column
  const int lo   = (lane < 16) ? 0 : 16;// A fragment K base (contiguous K = 16h + s, matches B)
  const int cb   = (lane < 16) ? 0 : 16;// B fragment K base (contiguous K = 16h + s, matches A)
  const int col  = lane & 15;

  // ---- preload all 18 A (weight) fragments into registers ----
  v16bf afrag[9][2];
#pragma unroll
  for (int t = 0; t < 9; ++t)
#pragma unroll
    for (int mt = 0; mt < 2; ++mt) {
      const __bf16* wp = wb + (size_t)(t * 32 + (mt * 16 + row)) * 32;
      v8bf a0 = *(const v8bf*)(wp + lo);        // K = lo..lo+7
      v8bf a1 = *(const v8bf*)(wp + lo + 8);    // K = lo+8..lo+15
      v16bf a;
#pragma unroll
      for (int i = 0; i < 8; ++i) { a[i] = a0[i]; a[i + 8] = a1[i]; }
      afrag[t][mt] = a;
    }

  const int w0 = wave * 16;                         // tile start along W
  float* obase = out + (size_t)n * C_ * cstride + (size_t)d * (H_ * W_);
  const int orow0 = (lane < 16) ? 0 : 8;            // C/D layout row base

#pragma unroll
  for (int hl = 0; hl < HT; ++hl) {
    v8f acc0 = {};   // out channels 0..15
    v8f acc1 = {};   // out channels 16..31
#pragma unroll
    for (int kh = 0; kh < 3; ++kh) {
      const int r = hl + kh;                        // LDS row (already includes -1 halo)
#pragma unroll
      for (int kw = 0; kw < 3; ++kw) {
        const int t = kh * 3 + kw;
        // B fragment: 16 consecutive channels at fixed (row, col) -> 2x ds_load_b128
        const v16bf b =
            *(const v16bf*)&xs[((r * LDSW) + (w0 + col + kw)) * C_ + cb];
        acc0 = __builtin_amdgcn_wmma_f32_16x16x32_bf16(
            false, afrag[t][0], false, b, (short)0, acc0, false, false);
        acc1 = __builtin_amdgcn_wmma_f32_16x16x32_bf16(
            false, afrag[t][1], false, b, (short)0, acc1, false, false);
      }
    }
    // ---- store D: lane holds col=lane&15, rows orow0..orow0+7 (+16 for acc1) ----
    float* op = obase + (size_t)(h0 + hl) * W_ + (w0 + col);
#pragma unroll
    for (int rr = 0; rr < 8; ++rr) {
      op[(size_t)(orow0 + rr)      * cstride] = acc0[rr];
      op[(size_t)(orow0 + rr + 16) * cstride] = acc1[rr];
    }
  }
}

// ---------------------------------------------------------------------------
extern "C" void kernel_launch(void* const* d_in, const int* in_sizes, int n_in,
                              void* d_out, int out_size, void* d_ws, size_t ws_size,
                              hipStream_t stream) {
  const float* x = (const float*)d_in[0];          // (8,32,16,160,160) f32
  const float* w = (const float*)d_in[1];          // (32,32,1,3,3) f32
  float* out = (float*)d_out;                      // (8,32,16,160,160) f32
  unsigned short* wb = (unsigned short*)d_ws;      // 9*32*32 bf16 = 18 KB

  bnn_prep_weights<<<4, 256, 0, stream>>>(w, wb);
  const int grid = N_ * D_ * (H_ / HT);            // 8*16*40 = 5120 blocks
  bnn_conv3d_wmma<<<grid, 320, 0, stream>>>(x, wb, out);
}
